// RealMambaBlock_64330020160111
// MI455X (gfx1250) — compile-verified, matrix operand order checked by reading, not run on hardware
//
#include <hip/hip_runtime.h>
#include <math.h>

typedef __attribute__((ext_vector_type(16))) __bf16 v16bf;
typedef __attribute__((ext_vector_type(8)))  __bf16 v8bf;
typedef __attribute__((ext_vector_type(4)))  __bf16 v4bf;
typedef __attribute__((ext_vector_type(8)))  float  v8f;

#define BATCH   2
#define SEQ     1024
#define D_MODEL 1024
#define D_INNER 2048
#define D_STATE 16
#define DT_RANK 64
#define NPROJ   96       // DT_RANK + 2*D_STATE
#define NPROJP  128      // padded proj width (tile-aligned)

// ---------------- Tiled bf16 WMMA GEMM ----------------
// C[M,N] = A[M,K] @ B[K,N], fp32 in/out, bf16 WMMA compute (f32 accum).
// Requirements (guaranteed by caller): M % 128 == 0, N % 128 == 0, K % 32 == 0.
// Block tile 128x128, K-step 32, double-buffered LDS, 8 waves.
// Wave tile 32x64: waves arranged 4 (M) x 2 (N); 8 WMMAs per wave per K-step.
#define BM 128
#define BN 128
#define BK 32
#define SP 40            // LDS pitch in bf16 (80B: 16B-aligned, bank-spread)

__global__ __launch_bounds__(256)
void wmma_gemm_bf16(const float* __restrict__ A, const float* __restrict__ B,
                    float* __restrict__ C, const float* __restrict__ bias,
                    int M, int N, int K, int lda, int ldb, int ldc, int epi)
{
  __shared__ __attribute__((aligned(16))) __bf16 sA[2][BM * SP];  // row-major rows
  __shared__ __attribute__((aligned(16))) __bf16 sB[2][BN * SP];  // col-major cols

  const int tid  = threadIdx.x;
  const int w    = tid >> 5;
  const int lane = tid & 31;
  const int half = lane >> 4;          // 16-bit fragment: lanes 16-31 hold k+8 runs
  const int l16  = lane & 15;
  const int wm   = w & 3;              // wave row group (0..3) -> rows wm*32..
  const int wn   = w >> 2;             // wave col group (0..1) -> cols wn*64..

  const int rowBase = blockIdx.y * BM;
  const int colBase = blockIdx.x * BN;

  // Stage one K-tile: gather ALL global loads into registers first (keeps
  // many loads in flight -> incremental s_wait_loadcnt), then convert+store.
  auto stage = [&](int kt, int buf) {
    float4 av[4];
#pragma unroll
    for (int j = 0; j < 4; ++j) {                     // A: 128x32 = 1024 float4
      int i4 = tid + 256 * j;
      int r  = i4 >> 3;                               // 0..127
      int k  = (i4 & 7) * 4;                          // 0,4,..,28
      av[j] = *(const float4*)&A[(size_t)(rowBase + r) * lda + kt + k];
    }
    float bv[4][4];
#pragma unroll
    for (int j = 0; j < 4; ++j) {                     // B: 4 k-consecutive of one col
      int i  = tid + 256 * j;
      int n  = i & 127;
      int kb = (i >> 7) * 4;
      const float* bp = &B[(size_t)(kt + kb) * ldb + colBase + n];
#pragma unroll
      for (int d = 0; d < 4; ++d) bv[j][d] = bp[(size_t)d * ldb];
    }
    // convert + LDS store (8B packed stores)
#pragma unroll
    for (int j = 0; j < 4; ++j) {
      int i4 = tid + 256 * j;
      int r  = i4 >> 3;
      int k  = (i4 & 7) * 4;
      v4bf pk = { (__bf16)av[j].x, (__bf16)av[j].y, (__bf16)av[j].z, (__bf16)av[j].w };
      *(v4bf*)&sA[buf][r * SP + k] = pk;
    }
#pragma unroll
    for (int j = 0; j < 4; ++j) {
      int i  = tid + 256 * j;
      int n  = i & 127;
      int kb = (i >> 7) * 4;
      v4bf pk = { (__bf16)bv[j][0], (__bf16)bv[j][1], (__bf16)bv[j][2], (__bf16)bv[j][3] };
      *(v4bf*)&sB[buf][n * SP + kb] = pk;
    }
  };

  v8f acc[2][4] = {};
  stage(0, 0);

  int buf = 0;
  for (int kt = 0; kt < K; kt += BK) {
    __syncthreads();
    if (kt + BK < K) stage(kt + BK, buf ^ 1);

    // A fragments (2 row tiles): per lane, k runs [8h,8h+8) and [16+8h,...)
    v16bf af[2];
#pragma unroll
    for (int mi = 0; mi < 2; ++mi) {
      const __bf16* arow = &sA[buf][(wm * 32 + mi * 16 + l16) * SP];
      v8bf lo = *(const v8bf*)(arow + 8 * half);
      v8bf hi = *(const v8bf*)(arow + 16 + 8 * half);
#pragma unroll
      for (int i = 0; i < 8; ++i) { af[mi][i] = lo[i]; af[mi][i + 8] = hi[i]; }
    }
    // B fragments (4 col tiles)
    v16bf bfr[4];
#pragma unroll
    for (int nb = 0; nb < 4; ++nb) {
      const __bf16* bcol = &sB[buf][(wn * 64 + nb * 16 + l16) * SP];
      v8bf lo = *(const v8bf*)(bcol + 8 * half);
      v8bf hi = *(const v8bf*)(bcol + 16 + 8 * half);
#pragma unroll
      for (int i = 0; i < 8; ++i) { bfr[nb][i] = lo[i]; bfr[nb][i + 8] = hi[i]; }
    }
#pragma unroll
    for (int mi = 0; mi < 2; ++mi)
#pragma unroll
      for (int nb = 0; nb < 4; ++nb)
        acc[mi][nb] = __builtin_amdgcn_wmma_f32_16x16x32_bf16(
            false, af[mi], false, bfr[nb], (short)0, acc[mi][nb], false, false);

    buf ^= 1;
  }

  // C/D layout: VGPR r -> row = r + 8*half, col = l16 (per 16-col tile)
#pragma unroll
  for (int mi = 0; mi < 2; ++mi) {
#pragma unroll
    for (int nb = 0; nb < 4; ++nb) {
      int col = colBase + wn * 64 + nb * 16 + l16;
      float b = (epi == 1) ? bias[col] : 0.f;
#pragma unroll
      for (int r = 0; r < 8; ++r) {
        int row = rowBase + wm * 32 + mi * 16 + r + 8 * half;
        float v = acc[mi][nb][r];
        if (epi == 1) {                 // dt = softplus(acc + b_dt[col])
          v += b;
          v = (v > 20.f) ? v : log1pf(__expf(v));
        }
        C[(size_t)row * ldc + col] = v;
      }
    }
  }
}

// ---------------- Pad W_xproj (2048 x 96) -> (2048 x 128), zero-filled ----------------
__global__ __launch_bounds__(256)
void pad_xproj_kernel(const float* __restrict__ Wp, float* __restrict__ Wpad)
{
  int i = blockIdx.x * 256 + threadIdx.x;   // D_INNER * NPROJP
  int r = i >> 7;
  int c = i & (NPROJP - 1);
  Wpad[i] = (c < NPROJ) ? Wp[r * NPROJ + c] : 0.f;
}

// ---------------- Depthwise causal conv (width 4) + bias + SiLU ----------------
__global__ __launch_bounds__(256)
void conv_silu_kernel(const float* __restrict__ xz, const float* __restrict__ cw,
                      const float* __restrict__ cb, float* __restrict__ xc)
{
  int idx = blockIdx.x * 256 + threadIdx.x;  // BATCH*SEQ*D_INNER (2^22)
  int e = idx & (D_INNER - 1);
  int t = (idx >> 11) & (SEQ - 1);
  int b = idx >> 21;
  const float* xrow = xz + (size_t)b * SEQ * (2 * D_INNER);
  float s = cb[e];
#pragma unroll
  for (int j = 0; j < 4; ++j) {
    int tt = t - 3 + j;
    if (tt >= 0) s += cw[e * 4 + j] * xrow[(size_t)tt * (2 * D_INNER) + e];
  }
  xc[idx] = s / (1.f + __expf(-s));          // silu
}

// ---------------- Selective scan ----------------
// One thread per (b, e, n); 16 lanes (one channel's states) reduce y via shuffles.
__global__ __launch_bounds__(256)
void scan_kernel(const float* __restrict__ xz, const float* __restrict__ xc,
                 const float* __restrict__ proj, const float* __restrict__ dt,
                 const float* __restrict__ A_log, const float* __restrict__ Dv,
                 float* __restrict__ y, int T)
{
  int gid = blockIdx.x * 256 + threadIdx.x;  // BATCH*D_INNER*D_STATE
  int n  = gid & (D_STATE - 1);
  int ch = gid >> 4;
  int e  = ch & (D_INNER - 1);
  int b  = ch >> 11;

  float Ae = -__expf(A_log[e * D_STATE + n]);
  float De = Dv[e];
  float h  = 0.f;

  const float* projb = proj + (size_t)b * T * NPROJP;
  const float* dtb   = dt   + (size_t)b * T * D_INNER;
  const float* xcb   = xc   + (size_t)b * T * D_INNER;
  const float* zb    = xz   + (size_t)b * T * (2 * D_INNER) + D_INNER;
  float*       yb    = y    + (size_t)b * T * D_INNER;

  for (int t = 0; t < T; ++t) {
    float dtv = dtb[(size_t)t * D_INNER + e];
    float xv  = xcb[(size_t)t * D_INNER + e];
    float Bn  = projb[(size_t)t * NPROJP + DT_RANK + n];
    float Cn  = projb[(size_t)t * NPROJP + DT_RANK + D_STATE + n];

    h = h * __expf(dtv * Ae) + dtv * xv * Bn;
    float p = h * Cn;
    p += __shfl_xor(p, 8, 16);
    p += __shfl_xor(p, 4, 16);
    p += __shfl_xor(p, 2, 16);
    p += __shfl_xor(p, 1, 16);
    if (n == 0) {
      float zv = zb[(size_t)t * (2 * D_INNER) + e];
      yb[(size_t)t * D_INNER + e] = (p + xv * De) * (zv / (1.f + __expf(-zv)));
    }
  }
}

// ---------------- Launch ----------------
extern "C" void kernel_launch(void* const* d_in, const int* in_sizes, int n_in,
                              void* d_out, int out_size, void* d_ws, size_t ws_size,
                              hipStream_t stream)
{
  (void)in_sizes; (void)n_in; (void)out_size; (void)ws_size;
  const float* x       = (const float*)d_in[0];
  const float* W_in    = (const float*)d_in[1];
  const float* conv_w  = (const float*)d_in[2];
  const float* conv_b  = (const float*)d_in[3];
  const float* W_xproj = (const float*)d_in[4];
  const float* W_dt    = (const float*)d_in[5];
  const float* b_dt    = (const float*)d_in[6];
  const float* A_log   = (const float*)d_in[7];
  const float* Dv      = (const float*)d_in[8];
  const float* W_out   = (const float*)d_in[9];
  float* out = (float*)d_out;

  // workspace layout (floats); ~86 MB total, resident in the 192 MB L2
  float* xz    = (float*)d_ws;                              // [B,T,2*D_INNER]
  float* xc    = xz    + (size_t)BATCH * SEQ * 2 * D_INNER; // [B,T,D_INNER]
  float* wxp   = xc    + (size_t)BATCH * SEQ * D_INNER;     // [D_INNER,128] padded W_xproj
  float* proj  = wxp   + (size_t)D_INNER * NPROJP;          // [B,T,128]
  float* dtb   = proj  + (size_t)BATCH * SEQ * NPROJP;      // [B,T,D_INNER]
  float* yb    = dtb   + (size_t)BATCH * SEQ * D_INNER;     // [B,T,D_INNER]

  const int M = BATCH * SEQ;                                // 2048

  // 0) pad W_xproj to 128 columns so every GEMM dim is tile-aligned
  pad_xproj_kernel<<<(D_INNER * NPROJP) / 256, 256, 0, stream>>>(W_xproj, wxp);

  // 1) xz = x @ W_in                 (2048 x 1024 x 4096)
  wmma_gemm_bf16<<<dim3((2 * D_INNER) / BN, M / BM), 256, 0, stream>>>(
      x, W_in, xz, nullptr, M, 2 * D_INNER, D_MODEL, D_MODEL, 2 * D_INNER, 2 * D_INNER, 0);

  // 2) xc = silu(depthwise_conv(xz[:, :, :D_INNER]) + conv_b)
  conv_silu_kernel<<<(BATCH * SEQ * D_INNER) / 256, 256, 0, stream>>>(
      xz, conv_w, conv_b, xc);

  // 3) proj = xc @ W_xproj_pad       (2048 x 2048 x 128)
  wmma_gemm_bf16<<<dim3(NPROJP / BN, M / BM), 256, 0, stream>>>(
      xc, wxp, proj, nullptr, M, NPROJP, D_INNER, D_INNER, NPROJP, NPROJP, 0);

  // 4) dt = softplus(proj[:, :64] @ W_dt + b_dt)   (2048 x 64 x 2048)
  wmma_gemm_bf16<<<dim3(D_INNER / BN, M / BM), 256, 0, stream>>>(
      proj, W_dt, dtb, b_dt, M, D_INNER, DT_RANK, NPROJP, D_INNER, D_INNER, 1);

  // 5) selective scan + D skip + silu(z) gate
  scan_kernel<<<(BATCH * D_INNER * D_STATE) / 256, 256, 0, stream>>>(
      xz, xc, proj, dtb, A_log, Dv, yb, SEQ);

  // 6) out = y @ W_out               (2048 x 2048 x 1024)
  wmma_gemm_bf16<<<dim3(D_MODEL / BN, M / BM), 256, 0, stream>>>(
      yb, W_out, out, nullptr, M, D_MODEL, D_INNER, D_INNER, D_MODEL, D_MODEL, 0);
}
